// TreeLSTM_20658792693767
// MI455X (gfx1250) — compile-verified, matrix operand order checked by reading, not run on hardware
//
#include <hip/hip_runtime.h>
#include <hip/hip_bf16.h>

#if defined(__has_builtin)
# if __has_builtin(__builtin_amdgcn_global_load_async_to_lds_b128)
#  define HAVE_ASYNC_LDS 1
# endif
# if __has_builtin(__builtin_amdgcn_s_wait_asynccnt)
#  define HAVE_WAIT_ASYNC 1
# endif
#endif
#ifndef HAVE_ASYNC_LDS
# define HAVE_ASYNC_LDS 0
#endif
#ifndef HAVE_WAIT_ASYNC
# define HAVE_WAIT_ASYNC 0
#endif

typedef __attribute__((ext_vector_type(16))) __bf16 v16bf;
typedef __attribute__((ext_vector_type(8)))  __bf16 v8bf;
typedef __attribute__((ext_vector_type(4)))  __bf16 v4bf;
typedef __attribute__((ext_vector_type(8)))  float  v8f;
typedef __attribute__((ext_vector_type(4)))  int    v4i;

#define DDIM   256
#define KT     32
#define NKT    (DDIM / KT)   // 8 K-blocks
#define NG     768           // gates GEMM width (3*D)
#define WPAD_G 36            // K-stride pad for W_gates LDS tiles (parent_update)
#define WPAD_F 40            // K-stride pad for W_f LDS tiles (forget_fc) -> 1280 chunks = 5/thread

__device__ __forceinline__ float sigmoidf(float x) { return 1.0f / (1.0f + __expf(-x)); }

__device__ __forceinline__ v16bf cat8(v8bf lo, v8bf hi) {
  return __builtin_shufflevector(lo, hi, 0,1,2,3,4,5,6,7,8,9,10,11,12,13,14,15);
}

// ---- async global->LDS 16-byte copy (gfx1250 path with sync fallback) ----
// Builtin signature (from hipcc diagnostic): (v4i AS1* src, v4i AS3* dst, Ii, Ii).
// AS1 shares representation with generic; AS3 address == low 32 bits of the
// generic pointer (CDNA5 LDS aperture).
__device__ __forceinline__ void cp16_g2l(__bf16* ldst, const __bf16* gsrc) {
#if HAVE_ASYNC_LDS
  typedef __attribute__((address_space(1))) v4i gv4i;
  typedef __attribute__((address_space(3))) v4i lv4i;
  gv4i* g = (gv4i*)(unsigned long long)(__SIZE_TYPE__)gsrc;
  lv4i* l = (lv4i*)(unsigned int)(__SIZE_TYPE__)ldst;
  __builtin_amdgcn_global_load_async_to_lds_b128(g, l, 0, 0);
#else
  *(uint4*)ldst = *(const uint4*)gsrc;
#endif
}
__device__ __forceinline__ void wait_async_0() {
#if HAVE_ASYNC_LDS
# if HAVE_WAIT_ASYNC
  __builtin_amdgcn_s_wait_asynccnt(0);
# else
  asm volatile("s_wait_asynccnt 0x0" ::: "memory");
# endif
#endif
}
__device__ __forceinline__ void wait_async_5() {
#if HAVE_ASYNC_LDS
# if HAVE_WAIT_ASYNC
  __builtin_amdgcn_s_wait_asynccnt(5);
# else
  asm volatile("s_wait_asynccnt 0x5" ::: "memory");
# endif
#endif
}

// A fragment (16x32 bf16, MxK) from row-major LDS tile Ash[16][DDIM].
// CDNA5 wave32 layout: lanes 0-15 -> K runs {k0..+7, k0+16..+23}; lanes 16-31 -> +8.
__device__ __forceinline__ v16bf load_a_frag(const __bf16* Ash, int lane, int k0) {
  int m  = lane & 15;
  int kb = k0 + ((lane & 16) ? 8 : 0);
  const __bf16* p = Ash + m * DDIM + kb;
  v8bf lo = *(const v8bf*)(p);
  v8bf hi = *(const v8bf*)(p + 16);
  return cat8(lo, hi);
}

// B fragment (32x16 bf16, KxN) from K-minor LDS tile Wsh[col][stride].
__device__ __forceinline__ v16bf load_b_frag(const __bf16* Wsh, int lane, int n0, int stride) {
  int n  = n0 + (lane & 15);
  int kb = (lane & 16) ? 8 : 0;
  const __bf16* p = Wsh + n * stride + kb;
  v4bf b0 = *(const v4bf*)(p);
  v4bf b1 = *(const v4bf*)(p + 4);
  v4bf b2 = *(const v4bf*)(p + 16);
  v4bf b3 = *(const v4bf*)(p + 20);
  return cat8(__builtin_shufflevector(b0, b1, 0,1,2,3,4,5,6,7),
              __builtin_shufflevector(b2, b3, 0,1,2,3,4,5,6,7));
}

__global__ void zero_f32(float4* p, int n4) {
  int i = blockIdx.x * blockDim.x + threadIdx.x;
  if (i < n4) p[i] = make_float4(0.f, 0.f, 0.f, 0.f);
}

// Pack W[K=256][ncols] f32 -> Wt[NKT][ncols][wpad] bf16 (K-minor per 32-row block).
__global__ void pack_w(const float* __restrict__ W, __bf16* __restrict__ Wt,
                       int ncols, int wpad) {
  int idx = blockIdx.x * blockDim.x + threadIdx.x;
  if (idx >= DDIM * ncols) return;
  int k = idx / ncols, c = idx % ncols;
  int kt = k / KT, kr = k % KT;
  Wt[((size_t)kt * ncols + c) * wpad + kr] = (__bf16)W[idx];
}

// Kernel 1: sorted segment_ids -> contiguous child range per parent (no atomics).
__global__ __launch_bounds__(64) void seg_sum_hs(const float* __restrict__ hs,
                                                 const int* __restrict__ seg,
                                                 float* __restrict__ hs_sum,
                                                 int E, int P) {
  int p = blockIdx.x;
  if (p >= P) return;
  int lo = 0, hi = E;
  while (lo < hi) { int mid = (lo + hi) >> 1; if (seg[mid] < p) lo = mid + 1; else hi = mid; }
  int start = lo;
  hi = E;
  while (lo < hi) { int mid = (lo + hi) >> 1; if (seg[mid] <= p) lo = mid + 1; else hi = mid; }
  int end = lo;
  int c = threadIdx.x * 4;
  float4 s = make_float4(0.f, 0.f, 0.f, 0.f);
  for (int e = start; e < end; ++e) {
    float4 v = *(const float4*)(hs + (size_t)e * DDIM + c);
    s.x += v.x; s.y += v.y; s.z += v.z; s.w += v.w;
  }
  *(float4*)(hs_sum + (size_t)p * DDIM + c) = s;
}

// Kernel 2: 32-edge M-tile: F = sigmoid(child_hs @ W_f + b_f) * child_cs,
// atomically segment-summed into fc. Double-buffered async W staging.
__global__ __launch_bounds__(256) void forget_fc(const float* __restrict__ hs,
                                                 const float* __restrict__ cs,
                                                 const int* __restrict__ seg,
                                                 const __bf16* __restrict__ Wft,
                                                 const float* __restrict__ bfv,
                                                 float* __restrict__ fc,
                                                 int E, int P) {
  __shared__ __align__(16) __bf16 Ash[32 * DDIM];                 // 16 KB
  __shared__ __align__(16) __bf16 Wsh[2][DDIM * WPAD_F];          // 2 x 20 KB
  __shared__ int sseg[32];
  int tid = threadIdx.x, lane = tid & 31, w = tid >> 5;
  int e0 = blockIdx.x * 32;

  for (int it = tid; it < 32 * (DDIM / 4); it += 256) {           // stage A as bf16
    int m = it / (DDIM / 4), c4 = it % (DDIM / 4);
    float4 v = make_float4(0.f, 0.f, 0.f, 0.f);
    if (e0 + m < E) v = *(const float4*)(hs + (size_t)(e0 + m) * DDIM + c4 * 4);
    __bf16* dst = Ash + m * DDIM + c4 * 4;
    dst[0] = (__bf16)v.x; dst[1] = (__bf16)v.y; dst[2] = (__bf16)v.z; dst[3] = (__bf16)v.w;
  }
  if (tid < 32) sseg[tid] = (e0 + tid < E) ? seg[e0 + tid] : 0;

  const int CH = DDIM * WPAD_F / 8;                               // 1280 16B chunks
  // prologue: async-stage K-block 0 into buffer 0 (exactly 5 chunks/thread)
#pragma unroll
  for (int j = 0; j < 5; ++j) {
    int idx = tid + j * 256;
    cp16_g2l(&Wsh[0][idx * 8], Wft + (size_t)idx * 8);
  }
  wait_async_0();
  __syncthreads();

  v8f acc[2][2] = {};
  for (int kt = 0; kt < NKT; ++kt) {
    int cur = kt & 1;
    if (kt + 1 < NKT) {                                           // prefetch next tile
#pragma unroll
      for (int j = 0; j < 5; ++j) {
        int idx = tid + j * 256;
        cp16_g2l(&Wsh[cur ^ 1][idx * 8],
                 Wft + (size_t)(kt + 1) * (CH * 8) + (size_t)idx * 8);
      }
      wait_async_5();                                             // tile kt landed
    } else {
      wait_async_0();
    }
    __syncthreads();

    const __bf16* Wb = Wsh[cur];
    v16bf a0 = load_a_frag(Ash, lane, kt * KT);
    v16bf a1 = load_a_frag(Ash + 16 * DDIM, lane, kt * KT);
    v16bf b0 = load_b_frag(Wb, lane, w * 16, WPAD_F);
    acc[0][0] = __builtin_amdgcn_wmma_f32_16x16x32_bf16(false, a0, false, b0, (short)0, acc[0][0], false, false);
    acc[1][0] = __builtin_amdgcn_wmma_f32_16x16x32_bf16(false, a1, false, b0, (short)0, acc[1][0], false, false);
    v16bf b1 = load_b_frag(Wb, lane, 128 + w * 16, WPAD_F);
    acc[0][1] = __builtin_amdgcn_wmma_f32_16x16x32_bf16(false, a0, false, b1, (short)0, acc[0][1], false, false);
    acc[1][1] = __builtin_amdgcn_wmma_f32_16x16x32_bf16(false, a1, false, b1, (short)0, acc[1][1], false, false);
    __syncthreads();                                              // before buffer reuse
  }

  int hi16 = lane >> 4, nl = lane & 15;
  for (int mt = 0; mt < 2; ++mt) {
    for (int t = 0; t < 2; ++t) {
      int col = ((t * 8 + w) << 4) + nl;
      float bias = bfv[col];
      for (int r = 0; r < 8; ++r) {
        int m = mt * 16 + r + hi16 * 8, e = e0 + m;
        if (e < E) {
          float fv  = sigmoidf(acc[mt][t][r] + bias);
          float val = fv * cs[(size_t)e * DDIM + col];
          unsafeAtomicAdd(&fc[(size_t)sseg[m] * DDIM + col], val);
        }
      }
    }
  }
}

// Kernel 3: 16-parent tile: gates = hs_sum @ W_gates + b (48 N-tiles, 6/wave),
// fused activations + c = i*g + fc, h = o*tanh(c). Async W staging.
__global__ __launch_bounds__(256) void parent_update(const float* __restrict__ hs_sum,
                                                     const float* __restrict__ fc,
                                                     const __bf16* __restrict__ Wgt,
                                                     const float* __restrict__ bg,
                                                     float* __restrict__ out_h,
                                                     float* __restrict__ out_c,
                                                     int P) {
  __shared__ __align__(16) __bf16 Ash[16 * DDIM];                 // 8 KB
  __shared__ __align__(16) __bf16 Wsh[NG * WPAD_G];               // 54 KB
  int tid = threadIdx.x, lane = tid & 31, w = tid >> 5;
  int p0 = blockIdx.x * 16;

  for (int it = tid; it < 16 * (DDIM / 4); it += 256) {           // stage hs_sum tile
    int m = it / (DDIM / 4), c4 = it % (DDIM / 4);
    float4 v = make_float4(0.f, 0.f, 0.f, 0.f);
    if (p0 + m < P) v = *(const float4*)(hs_sum + (size_t)(p0 + m) * DDIM + c4 * 4);
    __bf16* dst = Ash + m * DDIM + c4 * 4;
    dst[0] = (__bf16)v.x; dst[1] = (__bf16)v.y; dst[2] = (__bf16)v.z; dst[3] = (__bf16)v.w;
  }

  const int CH = NG * WPAD_G / 8;                                 // 3456 16B chunks
  v8f acc[6] = {};
  for (int kt = 0; kt < NKT; ++kt) {
    __syncthreads();                                              // tile consumers done
    for (int idx = tid; idx < CH; idx += 256)
      cp16_g2l(&Wsh[idx * 8], Wgt + (size_t)kt * (CH * 8) + (size_t)idx * 8);
    wait_async_0();
    __syncthreads();
    v16bf a = load_a_frag(Ash, lane, kt * KT);
#pragma unroll
    for (int t = 0; t < 6; ++t) {                                 // tiles t*8+w: i,i,o,o,g,g
      v16bf b = load_b_frag(Wsh, lane, (t * 8 + w) * 16, WPAD_G);
      acc[t] = __builtin_amdgcn_wmma_f32_16x16x32_bf16(false, a, false, b, (short)0, acc[t], false, false);
    }
  }

  int hi16 = lane >> 4, nl = lane & 15;
  for (int g = 0; g < 2; ++g) {                                   // acc[g]=i, acc[2+g]=o, acc[4+g]=g
    int col = ((g * 8 + w) << 4) + nl;
    float bi = bg[col], bo = bg[256 + col], bgg = bg[512 + col];
    for (int r = 0; r < 8; ++r) {
      int m = r + hi16 * 8, p = p0 + m;
      if (p < P) {
        float iv = sigmoidf(acc[g][r] + bi);
        float ov = sigmoidf(acc[2 + g][r] + bo);
        float gv = tanhf(acc[4 + g][r] + bgg);
        float cv = iv * gv + fc[(size_t)p * DDIM + col];
        float hv = ov * tanhf(cv);
        out_h[(size_t)p * DDIM + col] = hv;
        out_c[(size_t)p * DDIM + col] = cv;
      }
    }
  }
}

extern "C" void kernel_launch(void* const* d_in, const int* in_sizes, int n_in,
                              void* d_out, int out_size, void* d_ws, size_t ws_size,
                              hipStream_t stream) {
  const float* child_hs = (const float*)d_in[0];
  const float* child_cs = (const float*)d_in[1];
  const int*   seg      = (const int*)d_in[2];
  // d_in[3] = num_segments (device scalar; P derived from out_size instead)
  const float* Wg       = (const float*)d_in[4];
  const float* bg       = (const float*)d_in[5];
  const float* Wf       = (const float*)d_in[6];
  const float* bfv      = (const float*)d_in[7];

  int E = in_sizes[0] / DDIM;
  int P = out_size / (2 * DDIM);

  float*  hs_sum = (float*)d_ws;                                  // [P, D] f32
  float*  fc     = hs_sum + (size_t)P * DDIM;                     // [P, D] f32
  __bf16* Wgt    = (__bf16*)(fc + (size_t)P * DDIM);              // [8][768][36] bf16
  __bf16* Wft    = Wgt + (size_t)NKT * NG * WPAD_G;               // [8][256][40] bf16
  float*  out_h  = (float*)d_out;
  float*  out_c  = out_h + (size_t)P * DDIM;

  int n4 = (P * DDIM) / 4;
  zero_f32<<<(n4 + 255) / 256, 256, 0, stream>>>((float4*)fc, n4);
  pack_w<<<(DDIM * NG + 255) / 256, 256, 0, stream>>>(Wg, Wgt, NG, WPAD_G);
  pack_w<<<(DDIM * DDIM + 255) / 256, 256, 0, stream>>>(Wf, Wft, DDIM, WPAD_F);
  seg_sum_hs<<<P, 64, 0, stream>>>(child_hs, seg, hs_sum, E, P);
  forget_fc<<<(E + 31) / 32, 256, 0, stream>>>(child_hs, child_cs, seg, Wft, bfv, fc, E, P);
  parent_update<<<(P + 15) / 16, 256, 0, stream>>>(hs_sum, fc, Wgt, bg, out_h, out_c, P);
}
